// TransformerBlock_16389595202261
// MI455X (gfx1250) — matrix operand order repaired and checked by reading, not compiled, not run on hardware
//
#include <hip/hip_runtime.h>
#include <stdint.h>

// ---------------------------------------------------------------------------
// Problem constants (from the reference)
// ---------------------------------------------------------------------------
#define Bc   4
#define Sc   2048
#define Dc   1024
#define Hc   16
#define DKc  64
#define DFFc 4096
#define Mrows (Bc * Sc)        // 8192
#define EPSc 1e-6f

typedef __bf16 bf16;
typedef bf16  v16bf __attribute__((ext_vector_type(16)));
typedef bf16  bf16x8 __attribute__((ext_vector_type(8)));
typedef float v8f   __attribute__((ext_vector_type(8)));

// ---------------------------------------------------------------------------
// WMMA helpers (CDNA5 v_wmma_f32_16x16x32_bf16, wave32)
// ---------------------------------------------------------------------------
union V16U { v16bf v; bf16x8 h[2]; };

__device__ __forceinline__ v8f bwmma(v16bf a, v16bf b, v8f c) {
  return __builtin_amdgcn_wmma_f32_16x16x32_bf16(
      /*neg_a=*/false, a, /*neg_b=*/false, b,
      /*c_mod=*/(short)0, c, /*reuse_a=*/false, /*reuse_b=*/false);
}

// A fragment: 16x32 bf16, row-major source, lda elements.
// lane<16 : row = lane,    elems = K k0+0..15
// lane>=16: row = lane-16, elems = K k0+16..31
__device__ __forceinline__ v16bf load_a_frag(const bf16* __restrict__ A, int lda,
                                             int row0, int k0, int lane) {
  int r = lane & 15;
  int koff = (lane & 16) ? 16 : 0;
  const bf16* p = A + (size_t)(row0 + r) * (size_t)lda + (size_t)(k0 + koff);
  V16U f;
  f.h[0] = *(const bf16x8*)(p);
  f.h[1] = *(const bf16x8*)(p + 8);
  return f.v;
}

// B fragment: 32x16 bf16 from Bt stored [N][K] (K contiguous), ldb elements.
// lane<16 : col = lane,    elems = K k0+0..15
// lane>=16: col = lane-16, elems = K k0+16..31
__device__ __forceinline__ v16bf load_b_frag(const bf16* __restrict__ Bt, int ldb,
                                             int n0, int k0, int lane) {
  int c = lane & 15;
  int koff = (lane & 16) ? 16 : 0;
  const bf16* p = Bt + (size_t)(n0 + c) * (size_t)ldb + (size_t)(k0 + koff);
  V16U f;
  f.h[0] = *(const bf16x8*)(p);
  f.h[1] = *(const bf16x8*)(p + 8);
  return f.v;
}

__device__ __forceinline__ float grp_max16(float v) {
  v = fmaxf(v, __shfl_xor(v, 1));
  v = fmaxf(v, __shfl_xor(v, 2));
  v = fmaxf(v, __shfl_xor(v, 4));
  v = fmaxf(v, __shfl_xor(v, 8));
  return v;
}
__device__ __forceinline__ float grp_sum16(float v) {
  v += __shfl_xor(v, 1);
  v += __shfl_xor(v, 2);
  v += __shfl_xor(v, 4);
  v += __shfl_xor(v, 8);
  return v;
}

// ---------------------------------------------------------------------------
// Weight transpose + f32 -> bf16 convert:  W[K][N] f32  ->  Wt[N][K] bf16
// ---------------------------------------------------------------------------
__global__ void transpose_cvt_kernel(const float* __restrict__ W, bf16* __restrict__ Wt,
                                     int K, int N) {
  __shared__ float tile[32][33];
  int k0 = blockIdx.y * 32;
  int n0 = blockIdx.x * 32;
  for (int i = threadIdx.y; i < 32; i += 8)
    tile[i][threadIdx.x] = W[(size_t)(k0 + i) * N + (n0 + threadIdx.x)];
  __syncthreads();
  for (int i = threadIdx.y; i < 32; i += 8)
    Wt[(size_t)(n0 + i) * K + (k0 + threadIdx.x)] = (bf16)tile[threadIdx.x][i];
}

// ---------------------------------------------------------------------------
// LayerNorm (BERT4Rec flavor: unbiased std, eps added to std), f32 in, bf16 out
// one block (256 thr) per row of D=1024
// ---------------------------------------------------------------------------
__global__ void layernorm_kernel(const float* __restrict__ x,
                                 const float* __restrict__ gamma,
                                 const float* __restrict__ beta,
                                 bf16* __restrict__ out) {
  int row = blockIdx.x;
  const float* xr = x + (size_t)row * Dc;
  int lane = threadIdx.x & 31;
  int wid  = threadIdx.x >> 5;
  __shared__ float red[8];

  float s = 0.f;
  for (int i = threadIdx.x; i < Dc; i += 256) s += xr[i];
  s += __shfl_xor(s, 1); s += __shfl_xor(s, 2); s += __shfl_xor(s, 4);
  s += __shfl_xor(s, 8); s += __shfl_xor(s, 16);
  if (lane == 0) red[wid] = s;
  __syncthreads();
  float tot = 0.f;
  #pragma unroll
  for (int i = 0; i < 8; ++i) tot += red[i];
  float mean = tot * (1.0f / Dc);
  __syncthreads();

  float ss = 0.f;
  for (int i = threadIdx.x; i < Dc; i += 256) {
    float d = xr[i] - mean;
    ss += d * d;
  }
  ss += __shfl_xor(ss, 1); ss += __shfl_xor(ss, 2); ss += __shfl_xor(ss, 4);
  ss += __shfl_xor(ss, 8); ss += __shfl_xor(ss, 16);
  if (lane == 0) red[wid] = ss;
  __syncthreads();
  float tss = 0.f;
  #pragma unroll
  for (int i = 0; i < 8; ++i) tss += red[i];
  float var = tss * (1.0f / (Dc - 1));           // ddof = 1
  float inv = 1.0f / (sqrtf(var) + EPSc);        // eps on std, not var

  bf16* orow = out + (size_t)row * Dc;
  for (int i = threadIdx.x; i < Dc; i += 256) {
    float v = gamma[i] * (xr[i] - mean) * inv + beta[i];
    orow[i] = (bf16)v;
  }
}

// ---------------------------------------------------------------------------
// WMMA GEMM:  C[M][N] = A[M][K](bf16 rowmajor) x Wt[N][K](bf16) + bias, epilogue
// block = 256 thr = 8 waves; wave w -> 16x64 output strip; block tile 128M x 64N
// K-loop software double-buffered; __launch_bounds__(...,1) raises the VGPR
// budget so the ~130 live VGPRs never spill.
// MODE 0: store bf16 Q/K head layout  [b][h][s][dk]
// MODE 1: store bf16 V  transposed    [b][h][dk][s]
// MODE 2: store f32  rowmajor + residual (f32)
// MODE 3: store bf16 rowmajor, tanh-GELU applied
// ---------------------------------------------------------------------------
template <int MODE>
__global__ void __launch_bounds__(256, 1)
gemm_wmma_kernel(const bf16* __restrict__ A,
                 const bf16* __restrict__ Bt,
                 const float* __restrict__ bias,
                 const float* __restrict__ resid,
                 void* __restrict__ out,
                 int M, int N, int K) {
  int lane = threadIdx.x & 31;
  int wid  = threadIdx.x >> 5;
  int row0 = blockIdx.y * 128 + wid * 16;
  int n0   = blockIdx.x * 64;

  v8f z = {0.f, 0.f, 0.f, 0.f, 0.f, 0.f, 0.f, 0.f};
  v8f acc0 = z, acc1 = z, acc2 = z, acc3 = z;

  // prologue: fragments for k0 = 0
  v16bf a_c  = load_a_frag(A, K, row0, 0, lane);
  v16bf b_c0 = load_b_frag(Bt, K, n0 +  0, 0, lane);
  v16bf b_c1 = load_b_frag(Bt, K, n0 + 16, 0, lane);
  v16bf b_c2 = load_b_frag(Bt, K, n0 + 32, 0, lane);
  v16bf b_c3 = load_b_frag(Bt, K, n0 + 48, 0, lane);

  for (int k0 = 0; k0 < K; k0 += 32) {
    int kn = (k0 + 32 < K) ? (k0 + 32) : k0;   // last iter reloads (discarded)
    v16bf a_n  = load_a_frag(A, K, row0, kn, lane);
    v16bf b_n0 = load_b_frag(Bt, K, n0 +  0, kn, lane);
    v16bf b_n1 = load_b_frag(Bt, K, n0 + 16, kn, lane);
    v16bf b_n2 = load_b_frag(Bt, K, n0 + 32, kn, lane);
    v16bf b_n3 = load_b_frag(Bt, K, n0 + 48, kn, lane);

    acc0 = bwmma(a_c, b_c0, acc0);
    acc1 = bwmma(a_c, b_c1, acc1);
    acc2 = bwmma(a_c, b_c2, acc2);
    acc3 = bwmma(a_c, b_c3, acc3);

    a_c = a_n; b_c0 = b_n0; b_c1 = b_n1; b_c2 = b_n2; b_c3 = b_n3;
  }

  v8f acc[4] = {acc0, acc1, acc2, acc3};
  int rbase = (lane & 16) ? 8 : 0;
  int cl = lane & 15;

  #pragma unroll
  for (int t = 0; t < 4; ++t) {
    int col = n0 + t * 16 + cl;
    float bv = bias[col];
    #pragma unroll
    for (int i = 0; i < 8; ++i) {
      int row = row0 + rbase + i;
      float v = acc[t][i] + bv;
      if (MODE == 0) {                  // [b][h][s][dk]
        int b = row / Sc, s = row & (Sc - 1);
        int h = col >> 6, dk = col & 63;
        ((bf16*)out)[(((size_t)(b * Hc + h)) * Sc + s) * DKc + dk] = (bf16)v;
      } else if (MODE == 1) {           // [b][h][dk][s]
        int b = row / Sc, s = row & (Sc - 1);
        int h = col >> 6, dk = col & 63;
        ((bf16*)out)[(((size_t)(b * Hc + h)) * DKc + dk) * Sc + s] = (bf16)v;
      } else if (MODE == 2) {           // f32 + residual
        ((float*)out)[(size_t)row * N + col] = v + resid[(size_t)row * N + col];
      } else {                          // MODE 3: GELU -> bf16
        float g = 0.5f * v *
                  (1.0f + tanhf(0.7978845608028654f * (v + 0.044715f * v * v * v)));
        ((bf16*)out)[(size_t)row * N + col] = (bf16)g;
      }
    }
  }
}

// ---------------------------------------------------------------------------
// Flash-style attention. One wave per 16-query tile; 4 waves / block.
// q,k: [b][h][s][dk] bf16 ; vT: [b][h][dk][s] bf16 ; mask: (B,1,S,S) int32
// out: attn bf16 [b][s][h*64+dk]  (row-major [B*S][D], ready as next A matrix)
// Load batching is staggered to bound register pressure:
//   K frags + mask -> score WMMAs (K frags die) -> V frag loads overlap the
//   softmax VALU/exp work -> P transpose through LDS -> P.V WMMAs.
// __launch_bounds__(128, 1) keeps everything in registers (no scratch spills).
// ---------------------------------------------------------------------------
__global__ void __launch_bounds__(128, 1)
attention_kernel(const bf16* __restrict__ q,
                 const bf16* __restrict__ k,
                 const bf16* __restrict__ vT,
                 const int*  __restrict__ mask,
                 bf16* __restrict__ attn) {
  int lane = threadIdx.x & 31;
  int wid  = threadIdx.x >> 5;
  int tile = blockIdx.x * 4 + wid;         // global q-tile
  int qt = tile & 127;                     // S/16 = 128 tiles per (b,h)
  int bh = tile >> 7;                      // 0 .. B*H-1
  int h = bh & (Hc - 1);
  int b = bh >> 4;

  const bf16* qb = q  + (size_t)bh * Sc * DKc;
  const bf16* kb = k  + (size_t)bh * Sc * DKc;
  const bf16* vb = vT + (size_t)bh * DKc * Sc;
  const int*  mb = mask + (size_t)b * Sc * Sc;

  int q0 = qt * 16;
  int rbase = (lane & 16) ? 8 : 0;
  int cl = lane & 15;

  // per-wave LDS staging tile for P (16 rows x 32 cols, pitch 40 -> 16B aligned rows)
  __shared__ bf16 plds_all[4][16][40];
  bf16 (*plds)[40] = plds_all[wid];

  // Q fragments kept resident (K = dk, two 32-steps)
  v16bf qa0 = load_a_frag(qb, DKc, q0, 0,  lane);
  v16bf qa1 = load_a_frag(qb, DKc, q0, 32, lane);

  v8f z = {0.f, 0.f, 0.f, 0.f, 0.f, 0.f, 0.f, 0.f};
  v8f o0 = z, o1 = z, o2 = z, o3 = z;
  float mrow[8], lrow[8];
  #pragma unroll
  for (int i = 0; i < 8; ++i) { mrow[i] = -1e30f; lrow[i] = 0.f; }

  for (int kblk = 0; kblk < Sc; kblk += 32) {
    // ---- batch 1: K fragments + mask words -------------------------------
    v16bf kf0 = load_b_frag(kb, DKc, kblk,      0,  lane);
    v16bf kf1 = load_b_frag(kb, DKc, kblk,      32, lane);
    v16bf kf2 = load_b_frag(kb, DKc, kblk + 16, 0,  lane);
    v16bf kf3 = load_b_frag(kb, DKc, kblk + 16, 32, lane);
    int c0 = kblk + cl;
    int c1 = kblk + 16 + cl;
    int mv0[8], mv1[8];
    #pragma unroll
    for (int i = 0; i < 8; ++i) {
      const int* mp = mb + (size_t)(q0 + rbase + i) * Sc;
      mv0[i] = mp[c0];
      mv1[i] = mp[c1];
    }

    // ---- scores: two 16x16 tiles covering 32 keys, K = DK = 64
    v8f s0 = z, s1 = z;
    s0 = bwmma(qa0, kf0, s0);
    s0 = bwmma(qa1, kf1, s0);
    s1 = bwmma(qa0, kf2, s1);
    s1 = bwmma(qa1, kf3, s1);

    // ---- batch 2: V fragments; loads hide under the softmax work below ---
    v16bf vf0 = load_b_frag(vb, Sc,  0, kblk, lane);
    v16bf vf1 = load_b_frag(vb, Sc, 16, kblk, lane);
    v16bf vf2 = load_b_frag(vb, Sc, 32, kblk, lane);
    v16bf vf3 = load_b_frag(vb, Sc, 48, kblk, lane);

    float t0[8], t1[8];
    #pragma unroll
    for (int i = 0; i < 8; ++i) {
      float x0 = s0[i] * 0.125f;                 // 1/sqrt(64)
      float x1 = s1[i] * 0.125f;
      if (mv0[i] == 0) x0 = -1e9f;
      if (mv1[i] == 0) x1 = -1e9f;
      t0[i] = x0; t1[i] = x1;
    }

    // ---- online softmax update (row stats live across each 16-lane half)
    float alpha[8], mn[8];
    #pragma unroll
    for (int i = 0; i < 8; ++i) {
      float mx = grp_max16(fmaxf(t0[i], t1[i]));
      mn[i] = fmaxf(mrow[i], mx);
      alpha[i] = __expf(mrow[i] - mn[i]);
      mrow[i] = mn[i];
    }
    #pragma unroll
    for (int i = 0; i < 8; ++i) {
      float p0 = __expf(t0[i] - mn[i]);
      float p1 = __expf(t1[i] - mn[i]);
      float rs = grp_sum16(p0 + p1);
      lrow[i] = lrow[i] * alpha[i] + rs;
      o0[i] *= alpha[i]; o1[i] *= alpha[i];
      o2[i] *= alpha[i]; o3[i] *= alpha[i];
      // scatter P into per-wave LDS tile (D-frag -> row-major)
      plds[rbase + i][cl]      = (bf16)p0;
      plds[rbase + i][cl + 16] = (bf16)p1;
    }
    __asm__ volatile("" ::: "memory");   // keep LDS store->load order (HW DS is in-order per wave)
    v16bf pa = load_a_frag(&plds[0][0], 40, 0, 0, lane);
    __asm__ volatile("" ::: "memory");

    // ---- O += P (16x32) x V (32x64); V frags already resident
    o0 = bwmma(pa, vf0, o0);
    o1 = bwmma(pa, vf1, o1);
    o2 = bwmma(pa, vf2, o2);
    o3 = bwmma(pa, vf3, o3);
  }

  // ---- normalize + store to row-major [B*S][D] bf16
  v8f oarr[4] = {o0, o1, o2, o3};
  #pragma unroll
  for (int t = 0; t < 4; ++t) {
    int col = h * DKc + t * 16 + cl;
    #pragma unroll
    for (int i = 0; i < 8; ++i) {
      int row = q0 + rbase + i;
      float v = oarr[t][i] / lrow[i];
      attn[((size_t)b * Sc + row) * Dc + col] = (bf16)v;
    }
  }
}

// ---------------------------------------------------------------------------
// Host orchestration
// ---------------------------------------------------------------------------
extern "C" void kernel_launch(void* const* d_in, const int* in_sizes, int n_in,
                              void* d_out, int out_size, void* d_ws, size_t ws_size,
                              hipStream_t stream) {
  (void)in_sizes; (void)n_in; (void)out_size; (void)ws_size;
  const float* x     = (const float*)d_in[0];
  const int*   mask  = (const int*)  d_in[1];
  const float* ln1_a = (const float*)d_in[2];
  const float* ln1_b = (const float*)d_in[3];
  const float* Wq    = (const float*)d_in[4];
  const float* bq    = (const float*)d_in[5];
  const float* Wk    = (const float*)d_in[6];
  const float* bk    = (const float*)d_in[7];
  const float* Wv    = (const float*)d_in[8];
  const float* bv    = (const float*)d_in[9];
  const float* Wo    = (const float*)d_in[10];
  const float* bo    = (const float*)d_in[11];
  const float* ln2_a = (const float*)d_in[12];
  const float* ln2_b = (const float*)d_in[13];
  const float* W1    = (const float*)d_in[14];
  const float* b1    = (const float*)d_in[15];
  const float* W2    = (const float*)d_in[16];
  const float* b2    = (const float*)d_in[17];

  uint8_t* ws = (uint8_t*)d_ws;
  size_t off = 0;
  auto take = [&](size_t bytes) -> void* {
    void* p = ws + off;
    off += (bytes + 255) & ~(size_t)255;
    return p;
  };

  bf16* Wqt = (bf16*)take((size_t)Dc * Dc * 2);
  bf16* Wkt = (bf16*)take((size_t)Dc * Dc * 2);
  bf16* Wvt = (bf16*)take((size_t)Dc * Dc * 2);
  bf16* Wot = (bf16*)take((size_t)Dc * Dc * 2);
  bf16* W1t = (bf16*)take((size_t)Dc * DFFc * 2);
  bf16* W2t = (bf16*)take((size_t)DFFc * Dc * 2);
  bf16* xb   = (bf16*)take((size_t)Mrows * Dc * 2);
  bf16* qbuf = (bf16*)take((size_t)Mrows * Dc * 2);
  bf16* kbuf = (bf16*)take((size_t)Mrows * Dc * 2);
  bf16* vbuf = (bf16*)take((size_t)Mrows * Dc * 2);
  bf16* attn = (bf16*)take((size_t)Mrows * Dc * 2);
  float* x1  = (float*)take((size_t)Mrows * Dc * 4);
  bf16* xb2  = (bf16*)take((size_t)Mrows * Dc * 2);
  bf16* hbuf = (bf16*)take((size_t)Mrows * DFFc * 2);

  dim3 tb(32, 8, 1);
  // weight transpose+convert
  transpose_cvt_kernel<<<dim3(Dc / 32, Dc / 32), tb, 0, stream>>>(Wq, Wqt, Dc, Dc);
  transpose_cvt_kernel<<<dim3(Dc / 32, Dc / 32), tb, 0, stream>>>(Wk, Wkt, Dc, Dc);
  transpose_cvt_kernel<<<dim3(Dc / 32, Dc / 32), tb, 0, stream>>>(Wv, Wvt, Dc, Dc);
  transpose_cvt_kernel<<<dim3(Dc / 32, Dc / 32), tb, 0, stream>>>(Wo, Wot, Dc, Dc);
  transpose_cvt_kernel<<<dim3(DFFc / 32, Dc / 32), tb, 0, stream>>>(W1, W1t, Dc, DFFc);
  transpose_cvt_kernel<<<dim3(Dc / 32, DFFc / 32), tb, 0, stream>>>(W2, W2t, DFFc, Dc);

  // LN1
  layernorm_kernel<<<Mrows, 256, 0, stream>>>(x, ln1_a, ln1_b, xb);

  // QKV projections
  dim3 gD(Dc / 64, Mrows / 128);
  gemm_wmma_kernel<0><<<gD, 256, 0, stream>>>(xb, Wqt, bq, nullptr, qbuf, Mrows, Dc, Dc);
  gemm_wmma_kernel<0><<<gD, 256, 0, stream>>>(xb, Wkt, bk, nullptr, kbuf, Mrows, Dc, Dc);
  gemm_wmma_kernel<1><<<gD, 256, 0, stream>>>(xb, Wvt, bv, nullptr, vbuf, Mrows, Dc, Dc);

  // attention: B*H*S/16 = 8192 q-tiles, 4 waves per block
  attention_kernel<<<(Bc * Hc * (Sc / 16)) / 4, 128, 0, stream>>>(qbuf, kbuf, vbuf, mask, attn);

  // output projection + residual -> x1 (f32)
  gemm_wmma_kernel<2><<<gD, 256, 0, stream>>>(attn, Wot, bo, x, x1, Mrows, Dc, Dc);

  // LN2
  layernorm_kernel<<<Mrows, 256, 0, stream>>>(x1, ln2_a, ln2_b, xb2);

  // FFN1 (+GELU) -> bf16
  dim3 gF1(DFFc / 64, Mrows / 128);
  gemm_wmma_kernel<3><<<gF1, 256, 0, stream>>>(xb2, W1t, b1, nullptr, hbuf, Mrows, DFFc, Dc);

  // FFN2 + residual -> d_out (f32)
  gemm_wmma_kernel<2><<<gD, 256, 0, stream>>>(hbuf, W2t, b2, x1, (float*)d_out, Mrows, Dc, DFFc);
}
